// KoLeoLoss_42872363549176
// MI455X (gfx1250) — matrix operand order repaired and checked by reading, not compiled, not run on hardware
//
#include <hip/hip_runtime.h>

// ---------------------------------------------------------------------------
// KoLeo loss for MI455X (gfx1250, wave32, WMMA).
//   N = 16384 rows, D = 256. Gram matrix = 137 GFLOP; x(f16) = 8 MB, fully
//   L2-resident (192 MB L2). f16 WMMA (f32 accum) selects the nearest
//   neighbor; exact fp32 for norms, distances, and the final log-mean.
//
//   K2 design: register-blocked GEMM-like loop, no LDS. Each wave owns TWO
//   16-row M-tiles (A fragments for full K=256 live in 128 VGPRs) and one
//   2048-column chunk. Every 16-column N-tile costs 16 global b128 loads
//   (B fragments, shared by both M-tiles) feeding 16 v_wmma_f32_16x16x32_f16
//   on two independent accumulator chains. The compiler pipelines global
//   loads with s_clause + partial s_wait_loadcnt (verified in earlier rounds),
//   unlike the ds_load path which it serializes with s_wait_dscnt 0.
// ---------------------------------------------------------------------------

typedef __attribute__((ext_vector_type(16))) _Float16 v16h;
typedef __attribute__((ext_vector_type(8)))  float    v8f;

#define N_ROWS     16384
#define DIM        256
#define KOLEO_EPS  1e-8f
#define N_CHUNKS   8
#define CHUNK_COLS (N_ROWS / N_CHUNKS)   // 2048 columns per wave

union Frag16 {            // one WMMA 16-bit A/B fragment: 8 VGPRs / 32 bytes per lane
    v16h   h;
    float4 f4[2];
};

// ---------------------------------------------------------------------------
// K1: wave-per-row L2 normalization -> f16 normalized rows + fp32 inv norms.
// ---------------------------------------------------------------------------
__global__ void __launch_bounds__(256)
koleo_normalize(const float* __restrict__ in,
                _Float16* __restrict__ xh,
                float* __restrict__ invn) {
    int wave = (int)((blockIdx.x * blockDim.x + threadIdx.x) >> 5);
    int lane = (int)(threadIdx.x & 31);
    if (wave >= N_ROWS) return;

    const float4* row4 = (const float4*)(in + (size_t)wave * DIM);
    float4 v0 = row4[lane];
    float4 v1 = row4[lane + 32];

    float ss = v0.x * v0.x + v0.y * v0.y + v0.z * v0.z + v0.w * v0.w +
               v1.x * v1.x + v1.y * v1.y + v1.z * v1.z + v1.w * v1.w;
    #pragma unroll
    for (int off = 16; off; off >>= 1) ss += __shfl_xor(ss, off, 32);

    float scale = 1.0f / fmaxf(sqrtf(ss), KOLEO_EPS);   // x / max(||x||, eps)
    if (lane == 0) invn[wave] = scale;

    union { _Float16 h[4]; float2 f2; } o0, o1;
    o0.h[0] = (_Float16)(v0.x * scale); o0.h[1] = (_Float16)(v0.y * scale);
    o0.h[2] = (_Float16)(v0.z * scale); o0.h[3] = (_Float16)(v0.w * scale);
    o1.h[0] = (_Float16)(v1.x * scale); o1.h[1] = (_Float16)(v1.y * scale);
    o1.h[2] = (_Float16)(v1.z * scale); o1.h[3] = (_Float16)(v1.w * scale);

    float2* dst = (float2*)(xh + (size_t)wave * DIM);
    dst[lane]      = o0.f2;
    dst[lane + 32] = o1.f2;
}

// ---------------------------------------------------------------------------
// K2: Gram + running argmax, 2 M-tiles per wave.
//
// Fragment addressing (CDNA5 ISA):
//   A 16x32 f16 : lane (l16,half) holds row M=m0+l16,
//                 K = kb + half*8 + [0,8)  and  kb + 16 + half*8 + [0,8).
//   B 32x16 f16 : lane holds col N=n0+l16, same K split as A.
//   C 16x16 f32 : VGPR r -> (M = r + 8*half, N = n0 + l16).
// For X·X^T both fragments are contiguous 16B slices of row-major xh
// -> plain global b128 loads, no transpose needed.
// ---------------------------------------------------------------------------
__global__ void __launch_bounds__(256)
koleo_gram_argmax(const _Float16* __restrict__ xh,
                  float* __restrict__ pval,
                  int*   __restrict__ pidx) {
    const int wid   = (int)((blockIdx.x * blockDim.x + threadIdx.x) >> 5);
    const int lane  = (int)(threadIdx.x & 31);
    const int mpair = wid >> 3;          // 0..511 : pair of M-tiles
    const int chunk = wid & 7;           // 0..7   : column chunk
    const int m0a   = mpair * 32;        // first M-tile
    const int m0b   = m0a + 16;          // second M-tile
    const int half  = lane >> 4;
    const int l16   = lane & 15;

    // Register-resident A fragments for both M-tiles, full K=256 (128 VGPRs).
    const _Float16* arow0 = xh + (size_t)(m0a + l16) * DIM;
    const _Float16* arow1 = xh + (size_t)(m0b + l16) * DIM;
    Frag16 a0[8], a1[8];
    #pragma unroll
    for (int kc = 0; kc < 8; ++kc) {
        a0[kc].f4[0] = *(const float4*)(arow0 + kc * 32 + half * 8);
        a0[kc].f4[1] = *(const float4*)(arow0 + kc * 32 + 16 + half * 8);
        a1[kc].f4[0] = *(const float4*)(arow1 + kc * 32 + half * 8);
        a1[kc].f4[1] = *(const float4*)(arow1 + kc * 32 + 16 + half * 8);
    }

    float best[16];
    int   bidx[16];
    #pragma unroll
    for (int r = 0; r < 16; ++r) { best[r] = -2.0f; bidx[r] = 0; }

    const int n_start = chunk * CHUNK_COLS;
    for (int n0 = n_start; n0 < n_start + CHUNK_COLS; n0 += 16) {
        const _Float16* brow = xh + (size_t)(n0 + l16) * DIM + half * 8;

        v8f c0 = {};
        v8f c1 = {};
        #pragma unroll
        for (int kc = 0; kc < 8; ++kc) {
            Frag16 b;                                     // shared by both M-tiles
            b.f4[0] = *(const float4*)(brow + kc * 32);
            b.f4[1] = *(const float4*)(brow + kc * 32 + 16);
            c0 = __builtin_amdgcn_wmma_f32_16x16x32_f16(
                     false, a0[kc].h, false, b.h, (short)0, c0, false, false);
            c1 = __builtin_amdgcn_wmma_f32_16x16x32_f16(
                     false, a1[kc].h, false, b.h, (short)0, c1, false, false);
        }

        const int gN = n0 + l16;

        if (n0 == m0a) {                      // wave-uniform: diagonal in tile A
            #pragma unroll
            for (int r = 0; r < 8; ++r) {
                float v = c0[r];
                int gM = m0a + half * 8 + r;
                if (gM == gN) v = -1.0f;      // mask self
                if (v > best[r]) { best[r] = v; bidx[r] = gN; }
            }
        } else {
            #pragma unroll
            for (int r = 0; r < 8; ++r) {
                float v = c0[r];
                if (v > best[r]) { best[r] = v; bidx[r] = gN; }
            }
        }

        if (n0 == m0b) {                      // wave-uniform: diagonal in tile B
            #pragma unroll
            for (int r = 0; r < 8; ++r) {
                float v = c1[r];
                int gM = m0b + half * 8 + r;
                if (gM == gN) v = -1.0f;      // mask self
                if (v > best[8 + r]) { best[8 + r] = v; bidx[8 + r] = gN; }
            }
        } else {
            #pragma unroll
            for (int r = 0; r < 8; ++r) {
                float v = c1[r];
                if (v > best[8 + r]) { best[8 + r] = v; bidx[8 + r] = gN; }
            }
        }
    }

    // Reduce across the 16 lanes sharing each M (xor stays within a half).
    #pragma unroll
    for (int q = 0; q < 16; ++q) {
        float bv = best[q];
        int   bi = bidx[q];
        #pragma unroll
        for (int off = 8; off; off >>= 1) {
            float ov = __shfl_xor(bv, off, 32);
            int   oi = __shfl_xor(bi, off, 32);
            if (ov > bv || (ov == bv && oi < bi)) { bv = ov; bi = oi; }
        }
        if (l16 == 0) {
            int m0  = (q < 8) ? m0a : m0b;
            int row = m0 + half * 8 + (q & 7);
            pval[row * N_CHUNKS + chunk] = bv;
            pidx[row * N_CHUNKS + chunk] = bi;
        }
    }
}

// ---------------------------------------------------------------------------
// K3: wave-per-row. Merge the 8 chunk partials (argmax, tie -> lowest index),
//     then exact fp32 distance log(||x_i - x_j + eps||_2 + eps).
// ---------------------------------------------------------------------------
__global__ void __launch_bounds__(256)
koleo_dist(const float* __restrict__ in,
           const float* __restrict__ invn,
           const float* __restrict__ pval,
           const int*   __restrict__ pidx,
           float* __restrict__ logd) {
    int wave = (int)((blockIdx.x * blockDim.x + threadIdx.x) >> 5);
    int lane = (int)(threadIdx.x & 31);
    if (wave >= N_ROWS) return;

    float bv = -3.0f;
    int   bi = 0;
    if (lane < N_CHUNKS) {
        bv = pval[wave * N_CHUNKS + lane];
        bi = pidx[wave * N_CHUNKS + lane];
    }
    #pragma unroll
    for (int off = 4; off; off >>= 1) {   // width-8 xor: lanes 0..7 reduce
        float ov = __shfl_xor(bv, off, 8);
        int   oi = __shfl_xor(bi, off, 8);
        if (ov > bv || (ov == bv && oi < bi)) { bv = ov; bi = oi; }
    }
    int j = __shfl(bi, 0, 32);            // broadcast neighbor index

    float si = invn[wave];
    float sj = invn[j];
    const float* ri = in + (size_t)wave * DIM;
    const float* rj = in + (size_t)j    * DIM;

    float ss = 0.0f;
    #pragma unroll
    for (int k = lane; k < DIM; k += 32) {
        float d = ri[k] * si - rj[k] * sj + KOLEO_EPS;   // torch pairwise eps
        ss += d * d;
    }
    #pragma unroll
    for (int off = 16; off; off >>= 1) ss += __shfl_xor(ss, off, 32);

    if (lane == 0) logd[wave] = logf(sqrtf(ss) + KOLEO_EPS);
}

// ---------------------------------------------------------------------------
// K4: single-block deterministic reduction -> out = -mean(log distances).
// ---------------------------------------------------------------------------
__global__ void __launch_bounds__(256)
koleo_reduce(const float* __restrict__ logd, float* __restrict__ out) {
    __shared__ float smem[256];
    float acc = 0.0f;
    for (int i = (int)threadIdx.x; i < N_ROWS; i += 256) acc += logd[i];
    smem[threadIdx.x] = acc;
    __syncthreads();
    for (int off = 128; off; off >>= 1) {
        if ((int)threadIdx.x < off) smem[threadIdx.x] += smem[threadIdx.x + off];
        __syncthreads();
    }
    if (threadIdx.x == 0) out[0] = -smem[0] * (1.0f / (float)N_ROWS);
}

// ---------------------------------------------------------------------------
// Workspace layout (9.125 MB):
//   [0, 8MB)          : xh   - normalized rows, f16   (16384*256*2)
//   [8MB, 8.5MB)      : pval - per-chunk max values   (16384*8*4)
//   [8.5MB, 9MB)      : pidx - per-chunk argmax index (16384*8*4)
//   [9MB, 9MB+64KB)   : invn - inverse norms          (16384*4)
//   [.., +64KB)       : logd - per-row log distance   (16384*4)
// ---------------------------------------------------------------------------
extern "C" void kernel_launch(void* const* d_in, const int* in_sizes, int n_in,
                              void* d_out, int out_size, void* d_ws, size_t ws_size,
                              hipStream_t stream) {
    (void)in_sizes; (void)n_in; (void)out_size; (void)ws_size;

    const float* in  = (const float*)d_in[0];
    float*       out = (float*)d_out;
    char*        ws  = (char*)d_ws;

    _Float16* xh   = (_Float16*)ws;
    float*    pval = (float*)(ws + (size_t)8 * 1024 * 1024);
    int*      pidx = (int*)  (ws + (size_t)8 * 1024 * 1024 + 512 * 1024);
    float*    invn = (float*)(ws + (size_t)9 * 1024 * 1024);
    float*    logd = (float*)(ws + (size_t)9 * 1024 * 1024 + 64 * 1024);

    // K1: 16384 waves, 8 waves/block -> 2048 blocks.
    koleo_normalize<<<N_ROWS / 8, 256, 0, stream>>>(in, xh, invn);

    // K2: 512 M-pairs x 8 column chunks = 4096 waves -> 512 blocks.
    koleo_gram_argmax<<<(N_ROWS / 32) * N_CHUNKS / 8, 256, 0, stream>>>(xh, pval, pidx);

    // K3: 16384 waves -> 2048 blocks.
    koleo_dist<<<N_ROWS / 8, 256, 0, stream>>>(in, invn, pval, pidx, logd);

    // K4: single block, deterministic final reduction.
    koleo_reduce<<<1, 256, 0, stream>>>(logd, out);
}
